// Net_SDE_37821482009379
// MI455X (gfx1250) — matrix operand order repaired and checked by reading, not compiled, not run on hardware
//
#include <hip/hip_runtime.h>
#include <float.h>

typedef _Float16 f16;
typedef __attribute__((ext_vector_type(16))) _Float16 v16h;
typedef __attribute__((ext_vector_type(8)))  _Float16 v8h;
typedef __attribute__((ext_vector_type(2)))  _Float16 v2h;
typedef __attribute__((ext_vector_type(8)))  float    v8f;

#define NSTEPS 50
#define RATE   0.025f
#define NWAVES 8
#define PATHS_PER_BLOCK (16 * NWAVES)

struct NetP { const float* W[4]; const float* b[4]; };
struct AllP { NetP n[3]; };

// One-instruction ReLU: med3(x, 0, FLT_MAX) == clamp to [0, FLT_MAX].
__device__ __forceinline__ float relu(float x) {
  return __builtin_amdgcn_fmed3f(x, 0.f, FLT_MAX);
}

__device__ __forceinline__ v16h cat8(v8h a, v8h b) {
  v16h r;
#pragma unroll
  for (int i = 0; i < 8; ++i) { r[i] = a[i]; r[i + 8] = b[i]; }
  return r;
}

// A-fragment (16x32 f16) from row-major LDS tile H[16][64], k-chunk kc.
__device__ __forceinline__ v16h ldA(const f16* H, int kc, int m, int g) {
  const f16* p = H + m * 64 + kc * 32 + g * 8;
  return cat8(*(const v8h*)p, *(const v8h*)(p + 16));
}

// B-fragment (32x16 f16) from transposed LDS weights Wt[n][k] (row stride kp).
__device__ __forceinline__ v16h ldB(const f16* Wt, int kp, int kc, int nt, int m, int g) {
  const f16* p = Wt + (nt * 16 + m) * kp + kc * 32 + g * 8;
  return cat8(*(const v8h*)p, *(const v8h*)(p + 16));
}

// relu + packed f32->f16 convert + store one C tile column pair-wise.
__device__ __forceinline__ void stTile(f16* H, int nt, int m, int g, v8f c) {
#pragma unroll
  for (int r = 0; r < 8; r += 2) {
    v2h pk = __builtin_bit_cast(v2h, __builtin_amdgcn_cvt_pkrtz(relu(c[r]), relu(c[r + 1])));
    f16* q = H + (g * 8 + r) * 64 + nt * 16 + m;
    q[0]  = pk[0];
    q[64] = pk[1];   // next row (stride 64 f16)
  }
}

// Hidden layer: H <- relu(H(16x64) @ W(64x64 padded, bias in row k=50)), in place.
// Ones-column: Wt[50][50]==1 keeps H column 50 equal to 1.0 across layers.
__device__ __forceinline__ void hiddenLayer(f16* H, const f16* Wt, int m, int g) {
  v16h a0 = ldA(H, 0, m, g);
  v16h a1 = ldA(H, 1, m, g);
  asm volatile("" ::: "memory");
#pragma unroll
  for (int nt = 0; nt < 4; ++nt) {
    v8f c = {0.f, 0.f, 0.f, 0.f, 0.f, 0.f, 0.f, 0.f};
    v16h b0 = ldB(Wt, 64, 0, nt, m, g);
    c = __builtin_amdgcn_wmma_f32_16x16x32_f16(false, a0, false, b0, (short)0, c, false, false);
    v16h b1 = ldB(Wt, 64, 1, nt, m, g);
    c = __builtin_amdgcn_wmma_f32_16x16x32_f16(false, a1, false, b1, (short)0, c, false, false);
    stTile(H, nt, m, g, c);
  }
  asm volatile("" ::: "memory");
}

__global__ __launch_bounds__(256) void sde_sim(const float* __restrict__ zG,
                                               const float* __restrict__ z1G,
                                               AllP prm, float* __restrict__ gAcc, int MC) {
  __shared__ alignas(16) f16 sW1[3][64 * 32];   // [n][k] transposed; bias in k=3; W1[50][3]=1
  __shared__ alignas(16) f16 sW2[3][64 * 64];   // bias in k=50; W[50][50]=1
  __shared__ alignas(16) f16 sW3[3][64 * 64];   // bias in k=50; W[50][50]=1
  __shared__ alignas(16) f16 sW4[3][16 * 64];   // row n=0 only; bias in k=50
  __shared__ alignas(16) f16 sH[NWAVES][16 * 64];
  __shared__ float sY[NWAVES][16];
  __shared__ float sAcc[NSTEPS * 10];

  const int tid = threadIdx.x;

  // ---- zero weight pads + block accumulators ----
  for (int i = tid; i < 3 * 64 * 32; i += 256) ((f16*)sW1)[i] = (f16)0.f;
  for (int i = tid; i < 3 * 64 * 64; i += 256) ((f16*)sW2)[i] = (f16)0.f;
  for (int i = tid; i < 3 * 64 * 64; i += 256) ((f16*)sW3)[i] = (f16)0.f;
  for (int i = tid; i < 3 * 16 * 64; i += 256) ((f16*)sW4)[i] = (f16)0.f;
  for (int i = tid; i < NSTEPS * 10; i += 256) sAcc[i] = 0.f;
  __syncthreads();

  // ---- stage weights: f16, transposed [n][k], biases folded as extra K rows ----
  for (int net = 0; net < 3; ++net) {
    const NetP& N = prm.n[net];
    for (int i = tid; i < 3 * 50; i += 256) {
      int k = i / 50, n = i % 50;
      sW1[net][n * 32 + k] = (f16)N.W[0][i];
    }
    for (int i = tid; i < 50 * 50; i += 256) {
      int k = i / 50, n = i % 50;
      sW2[net][n * 64 + k] = (f16)N.W[1][i];
      sW3[net][n * 64 + k] = (f16)N.W[2][i];
    }
    for (int i = tid; i < 50; i += 256) {
      sW4[net][i]           = (f16)N.W[3][i];   // W3 is [50,1] -> row n=0, col k
      sW1[net][i * 32 + 3]  = (f16)N.b[0][i];   // bias via ones-input (A1[3]=1)
      sW2[net][i * 64 + 50] = (f16)N.b[1][i];   // bias via ones-column (H[:,50]=1)
      sW3[net][i * 64 + 50] = (f16)N.b[2][i];
    }
    if (tid == 0) {
      sW1[net][50 * 32 + 3]  = (f16)1.f;  // establish ones-column in layer-1 output
      sW2[net][50 * 64 + 50] = (f16)1.f;  // propagate ones-column
      sW3[net][50 * 64 + 50] = (f16)1.f;
      sW4[net][50]           = (f16)N.b[3][0];  // output bias
    }
  }
  __syncthreads();

  const int lane = tid & 31;
  const int wid  = tid >> 5;
  const int m    = lane & 15;
  const int g    = lane >> 4;
  f16*   H = sH[wid];
  float* Y = sY[wid];

  const int  pbase = ((int)blockIdx.x * NWAVES + wid) * 16;
  const int  p     = pbase + m;
  const bool valid = (lane < 16) && (p < MC);
  const int  pcl   = (p < MC) ? p : (MC - 1);
  const int  zoff  = pcl * NSTEPS;

  const float h   = 1.0f / NSTEPS;
  const float sqh = sqrtf(h);
  float S = 100.0f, V = 0.04f;

  const float Kc[5] = {90.f, 95.f, 100.f, 105.f, 110.f};
  const float Kp[5] = {70.f, 75.f, 80.f, 85.f, 90.f};

  // ---- hoist loop-invariant fragments into registers ----
  v16h B1f[3][4];   // layer-1 B fragments (K padded to 32, bias row k=3)
  v16h B4f[3][2];   // layer-4 B fragments (bias row k=50)
#pragma unroll
  for (int net = 0; net < 3; ++net) {
#pragma unroll
    for (int nt = 0; nt < 4; ++nt) B1f[net][nt] = ldB(sW1[net], 32, 0, nt, m, g);
    B4f[net][0] = ldB(sW4[net], 64, 0, 0, m, g);
    B4f[net][1] = ldB(sW4[net], 64, 1, 0, m, g);
  }

  for (int s = 0; s < NSTEPS; ++s) {
    const float t   = s * h;
    const float zv  = valid ? zG[zoff + s]  : 0.f;
    const float z1v = valid ? z1G[zoff + s] : 0.f;

    // X = [t, S, V, 1] padded to 16x32, built directly in A-fragment layout.
    v16h A1;
#pragma unroll
    for (int i = 0; i < 16; ++i) A1[i] = (f16)0.f;
    if (lane < 16) { A1[0] = (f16)t; A1[1] = (f16)S; A1[2] = (f16)V; A1[3] = (f16)1.f; }

    float y[3];
#pragma unroll
    for (int net = 0; net < 3; ++net) {
      // ---- layer 1: (16x32) @ (32x64), B from registers, bias folded ----
#pragma unroll
      for (int nt = 0; nt < 4; ++nt) {
        v8f c = {0.f, 0.f, 0.f, 0.f, 0.f, 0.f, 0.f, 0.f};
        c = __builtin_amdgcn_wmma_f32_16x16x32_f16(false, A1, false, B1f[net][nt],
                                                   (short)0, c, false, false);
        stTile(H, nt, m, g, c);
      }
      asm volatile("" ::: "memory");

      // ---- layers 2 & 3: (16x64) @ (64x64), in place, B from LDS ----
      hiddenLayer(H, sW2[net], m, g);
      hiddenLayer(H, sW3[net], m, g);

      // ---- layer 4: (16x64) @ (64x1 padded to 16), B from registers ----
      {
        v16h a0 = ldA(H, 0, m, g);
        v16h a1 = ldA(H, 1, m, g);
        v8f  c  = {0.f, 0.f, 0.f, 0.f, 0.f, 0.f, 0.f, 0.f};
        c = __builtin_amdgcn_wmma_f32_16x16x32_f16(false, a0, false, B4f[net][0],
                                                   (short)0, c, false, false);
        c = __builtin_amdgcn_wmma_f32_16x16x32_f16(false, a1, false, B4f[net][1],
                                                   (short)0, c, false, false);
        if (m == 0) {                     // lanes 0 and 16 hold column n=0
#pragma unroll
          for (int r = 0; r < 8; ++r) Y[g * 8 + r] = c[r];
        }
        asm volatile("" ::: "memory");
        y[net] = (lane < 16) ? Y[m] : 0.f;
      }
    }

    // Euler-Maruyama update
    const float dW = sqh * zv, dW1 = sqh * z1v;
    float Sn = relu(S + S * (RATE * h) + y[0] * dW);
    V = V + y[1] * h + y[2] * dW1;
    S = Sn;

    // payoffs (reference's "put" leg is also max(S-K,0))
    float pay[10];
#pragma unroll
    for (int j = 0; j < 5; ++j) {
      pay[j]     = valid ? relu(S - Kc[j]) : 0.f;
      pay[5 + j] = valid ? relu(S - Kp[j]) : 0.f;
    }
#pragma unroll
    for (int j = 0; j < 10; ++j) {
      float v = pay[j];
      v += __shfl_xor(v, 1, 32);
      v += __shfl_xor(v, 2, 32);
      v += __shfl_xor(v, 4, 32);
      v += __shfl_xor(v, 8, 32);
      pay[j] = v;
    }
    if (lane == 0) {
#pragma unroll
      for (int j = 0; j < 10; ++j) atomicAdd(&sAcc[s * 10 + j], pay[j]);
    }
  }

  __syncthreads();
  for (int i = tid; i < NSTEPS * 10; i += 256) atomicAdd(&gAcc[i], sAcc[i]);
}

__global__ void zeroK(float* __restrict__ acc) {
  int t = threadIdx.x + blockIdx.x * blockDim.x;
  if (t < NSTEPS * 10) acc[t] = 0.f;
}

__global__ void finishK(const int* __restrict__ idx, const float* __restrict__ acc,
                        float* __restrict__ out, int MC) {
  int t = threadIdx.x;
  if (t < 50) {
    int r = t / 5, j = t % 5;        // rows 0-4 calls, 5-9 puts
    int e = idx[r % 5];              // maturity step (1-based)
    float disc = __expf(-RATE * (float)e / (float)NSTEPS);
    float v = acc[(e - 1) * 10 + ((r >= 5) ? 5 : 0) + j];
    out[t] = v * disc / (float)MC;
  }
}

extern "C" void kernel_launch(void* const* d_in, const int* in_sizes, int n_in,
                              void* d_out, int out_size, void* d_ws, size_t ws_size,
                              hipStream_t stream) {
  const int*   indices = (const int*)d_in[0];
  const float* z       = (const float*)d_in[1];
  const float* z1      = (const float*)d_in[2];
  const int    MC      = in_sizes[1] / NSTEPS;

  AllP prm;
  for (int i = 0; i < 3; ++i)
    for (int l = 0; l < 4; ++l) {
      prm.n[i].W[l] = (const float*)d_in[4 + i * 8 + l * 2];
      prm.n[i].b[l] = (const float*)d_in[4 + i * 8 + l * 2 + 1];
    }

  float* gAcc = (float*)d_ws;
  float* out  = (float*)d_out;

  zeroK<<<1, 512, 0, stream>>>(gAcc);
  int blocks = (MC + PATHS_PER_BLOCK - 1) / PATHS_PER_BLOCK;
  sde_sim<<<blocks, 256, 0, stream>>>(z, z1, prm, gAcc, MC);
  finishK<<<1, 64, 0, stream>>>(indices, gAcc, out, MC);
}
